// Cross_Attention_24876450578626
// MI455X (gfx1250) — matrix operand order repaired and checked by reading, not compiled, not run on hardware
//
#include <hip/hip_runtime.h>
#include <hip/hip_bf16.h>

// ---------------------------------------------------------------------------
// Cross-attention for MI455X (gfx1250, wave32, WMMA).
// GEMMs: LDS double-buffered tiles filled via async global->LDS copies
// (ASYNCcnt / s_wait_asynccnt), consumed by v_wmma_f32_16x16x32_bf16.
// All LDS traffic goes through address_space(3) pointers so it lowers to
// ds_load/ds_store (DScnt), not flat ops.
// ---------------------------------------------------------------------------

typedef __attribute__((ext_vector_type(16))) __bf16 v16bf;
typedef __attribute__((ext_vector_type(8)))  __bf16 v8bf;
typedef __attribute__((ext_vector_type(8)))  float  v8f;
typedef int v4i __attribute__((vector_size(16)));

#define AS3 __attribute__((address_space(3)))
typedef AS3 char   lchar;
typedef AS3 float  lfloat;
typedef AS3 __bf16 lbf16;

#define B_DIM 8
#define T_DIM 1024
#define I_DIM 576
#define C_DIM 1024
#define H_DIM 16
#define D_DIM 64

// ---------- CDNA5 async global->LDS copy (guarded) --------------------------
#if defined(__AMDGCN__) && __has_builtin(__builtin_amdgcn_global_load_async_to_lds_b128) && \
    __has_builtin(__builtin_amdgcn_s_wait_asynccnt)
#define ASYNC_OK 1
#endif

static __device__ __forceinline__ void async_cp16(lchar* l, const __bf16* g) {
#ifdef ASYNC_OK
  __builtin_amdgcn_global_load_async_to_lds_b128(
      (__attribute__((address_space(1))) v4i*)(g), (AS3 v4i*)(l), 0, 0);
#else
  *(AS3 v8bf*)l = *(const v8bf*)g;   // synchronous fallback (ds_store)
#endif
}
static __device__ __forceinline__ void async_wait0() {
#ifdef ASYNC_OK
  __builtin_amdgcn_s_wait_asynccnt(0);
#endif
}

// ---- WMMA fragment helpers (16-bit layouts, contiguous k order, wave32) -------
// A (16x32 bf16): lane m=lane&15 holds row m; lo half K=k..k+15,
// hi half K=k+16..k+31 -> two 16B chunks.
// B (32x16 bf16) column-major: lane n holds column n; lo half K=k..k+15,
// hi half K=k+16..k+31 -> one contiguous 32B chunk.
static __device__ __forceinline__ v8f wmma_bf16(v16bf a, v16bf b, v8f c) {
  return __builtin_amdgcn_wmma_f32_16x16x32_bf16(
      false, a, false, b, (short)0, c, false, false);
}

// Fragments from an LDS tile with 64-element (128 B) row stride.
static __device__ __forceinline__ v16bf lds_a_frag(const lchar* lA, int mrow,
                                                   int kk, int lane) {
  int m = lane & 15, hi = lane >> 4;
  const lbf16* p = (const lbf16*)(lA + (mrow + m) * 128) + kk + hi * 16;
  union { v16bf v; v8bf h[2]; } u;
  u.h[0] = *(const AS3 v8bf*)p;
  u.h[1] = *(const AS3 v8bf*)(p + 8);
  return u.v;
}
static __device__ __forceinline__ v16bf lds_b_frag(const lchar* lB, int nrow,
                                                   int kk, int lane) {
  int n = lane & 15, hi = lane >> 4;
  return *(const AS3 v16bf*)((const lbf16*)(lB + (nrow + n) * 128) + kk + hi * 16);
}

// Fragments straight from global (row-major, stride lda) — used by attention.
static __device__ __forceinline__ v16bf load_a_frag(const __bf16* A, int lda,
                                                    int k0, int lane) {
  int m = lane & 15, hi = lane >> 4;
  const __bf16* p = A + (size_t)m * lda + k0 + hi * 16;
  union { v16bf v; v8bf h[2]; } u;
  u.h[0] = *(const v8bf*)p;
  u.h[1] = *(const v8bf*)(p + 8);
  return u.v;
}
static __device__ __forceinline__ v16bf load_b_frag(const __bf16* Wn0, int ldw,
                                                    int k0, int lane) {
  int n = lane & 15, hi = lane >> 4;
  return *(const v16bf*)(Wn0 + (size_t)n * ldw + k0 + hi * 16);
}

// ---------- LDS-tiled, double-buffered GEMM core -----------------------------
// Block = 256 threads (8 waves, wave32) computes 64M x 256N.
// Wave (wm=wave&1, wn=wave>>1) computes 32M x 64N (acc[2][4]).
// k-chunk = 64. Tiles: A 64x64 bf16 (8 KB), B 256x64 bf16 (32 KB).
#define LDS_BUF_BYTES  (8192 + 32768)          // 40960 per buffer
#define GEMM_LDS_BYTES (2 * LDS_BUF_BYTES)     // 81920 total

static __device__ __forceinline__ void stage_tiles(const __bf16* Arow0,
                                                   const __bf16* Wn0, int k0,
                                                   lchar* ls, int bufoff, int tid) {
  // A tile: 64 rows x 64 k = 512 x 16B chunks, 2 per thread.
#pragma unroll
  for (int s = 0; s < 2; ++s) {
    int idx = s * 256 + tid, r = idx >> 3, c = idx & 7;
    async_cp16(ls + bufoff + idx * 16, Arow0 + (size_t)r * C_DIM + k0 + c * 8);
  }
  // B tile: 256 rows x 64 k = 2048 x 16B chunks, 8 per thread.
#pragma unroll
  for (int s = 0; s < 8; ++s) {
    int idx = s * 256 + tid, r = idx >> 3, c = idx & 7;
    async_cp16(ls + bufoff + 8192 + idx * 16,
               Wn0 + (size_t)r * C_DIM + k0 + c * 8);
  }
}

template <int KDIM>
static __device__ __forceinline__ void gemm_block(const __bf16* Arow0,
                                                  const __bf16* Wn0,
                                                  lchar* ls, v8f acc[2][4]) {
  int tid = threadIdx.x, lane = tid & 31, wave = tid >> 5;
  int wm = wave & 1, wn = wave >> 1;
  constexpr int nk = KDIM / 64;

  stage_tiles(Arow0, Wn0, 0, ls, 0, tid);
  for (int kc = 0; kc < nk; ++kc) {
    int bo = (kc & 1) * LDS_BUF_BYTES;
    async_wait0();        // our chunk-kc copies have landed in LDS
    __syncthreads();      // everyone's chunk kc ready; prev buffer free
    if (kc + 1 < nk)
      stage_tiles(Arow0, Wn0, (kc + 1) * 64, ls, bo ^ LDS_BUF_BYTES, tid);
    const lchar* lA = ls + bo;
    const lchar* lB = ls + bo + 8192;
#pragma unroll
    for (int kk = 0; kk < 64; kk += 32) {
      v16bf a0 = lds_a_frag(lA, wm * 32,      kk, lane);
      v16bf a1 = lds_a_frag(lA, wm * 32 + 16, kk, lane);
#pragma unroll
      for (int j = 0; j < 4; ++j) {
        v16bf b = lds_b_frag(lB, wn * 64 + j * 16, kk, lane);
        acc[0][j] = wmma_bf16(a0, b, acc[0][j]);
        acc[1][j] = wmma_bf16(a1, b, acc[1][j]);
      }
    }
  }
}

// ---- fp32 -> bf16 conversion ----------------------------------------------
__global__ void __launch_bounds__(256) cvt_bf16(const float* __restrict__ in,
                                                __bf16* __restrict__ out, int n) {
  int i = blockIdx.x * 256 + threadIdx.x;
  if (i < n) out[i] = (__bf16)in[i];
}

// ---- GEMM 1: Q = (X * Wq^T + bq) * 1/sqrt(D) ------------------------------
__global__ void __launch_bounds__(256) gemm_q(const __bf16* __restrict__ X,
                                              const __bf16* __restrict__ Wq,
                                              const float* __restrict__ bq,
                                              __bf16* __restrict__ Q) {
  extern __shared__ __align__(16) char smem[];
  lchar* ls = (lchar*)smem;
  int lane = threadIdx.x & 31, wave = threadIdx.x >> 5;
  int wm = wave & 1, wn = wave >> 1;
  int row0 = blockIdx.x * 64, n0 = blockIdx.y * 256;
  v8f acc[2][4] = {};
  gemm_block<C_DIM>(X + (size_t)row0 * C_DIM, Wq + (size_t)n0 * C_DIM, ls, acc);
  int m = lane & 15, hi = lane >> 4;
#pragma unroll
  for (int mi = 0; mi < 2; ++mi)
#pragma unroll
    for (int j = 0; j < 4; ++j) {
      int col = n0 + wn * 64 + j * 16 + m;
      int rowb = row0 + wm * 32 + mi * 16;
      float bv = bq[col];
#pragma unroll
      for (int r = 0; r < 8; ++r)
        Q[(size_t)(rowb + r + 8 * hi) * C_DIM + col] =
            (__bf16)((acc[mi][j][r] + bv) * 0.125f);
    }
}

// ---- GEMM 2: KV = enc * Wkv^T + bkv ---------------------------------------
// K half -> row-major (b,i,c); V half -> VT[b][h][d][i] (pre-transposed).
__global__ void __launch_bounds__(256) gemm_kv(const __bf16* __restrict__ E,
                                               const __bf16* __restrict__ Wkv,
                                               const float* __restrict__ bkv,
                                               __bf16* __restrict__ Kb,
                                               __bf16* __restrict__ VT) {
  extern __shared__ __align__(16) char smem[];
  lchar* ls = (lchar*)smem;
  int lane = threadIdx.x & 31, wave = threadIdx.x >> 5;
  int wm = wave & 1, wn = wave >> 1;
  int row0 = blockIdx.x * 64, n0 = blockIdx.y * 256;
  v8f acc[2][4] = {};
  gemm_block<C_DIM>(E + (size_t)row0 * C_DIM, Wkv + (size_t)n0 * C_DIM, ls, acc);
  int m = lane & 15, hi = lane >> 4;
#pragma unroll
  for (int mi = 0; mi < 2; ++mi)
#pragma unroll
    for (int j = 0; j < 4; ++j) {
      int col = n0 + wn * 64 + j * 16 + m;
      int rowb = row0 + wm * 32 + mi * 16;
      float bv = bkv[col];
#pragma unroll
      for (int r = 0; r < 8; ++r) {
        int row = rowb + r + 8 * hi;           // row = b*I + i
        int b = row / I_DIM, i = row - b * I_DIM;
        float v = acc[mi][j][r] + bv;
        if (col < C_DIM) {
          Kb[((size_t)b * I_DIM + i) * C_DIM + col] = (__bf16)v;
        } else {
          int c = col - C_DIM, h = c >> 6, d = c & 63;
          VT[(((size_t)b * H_DIM + h) * D_DIM + d) * I_DIM + i] = (__bf16)v;
        }
      }
    }
}

// ---- Attention: per wave one (b, h, 16-row T tile) ------------------------
#define ATTN_WAVE_LDS (16 * I_DIM * 4)   // 36864 B per wave
#define ATTN_ROW_BYTES (I_DIM * 4)       // 2304 B per row (16B aligned)

__global__ void __launch_bounds__(64) attn(const __bf16* __restrict__ Q,
                                           const __bf16* __restrict__ Kb,
                                           const __bf16* __restrict__ VT,
                                           __bf16* __restrict__ Y) {
  extern __shared__ __align__(16) char smem[];
  int lane = threadIdx.x & 31, wave = threadIdx.x >> 5;
  int gw = blockIdx.x * 2 + wave;              // 8192 wave tiles total
  int tt = gw & 63, bh = gw >> 6;
  int h = bh & 15, b = bh >> 4;
  int t0 = tt * 16;

  lchar* wbase = (lchar*)smem + wave * ATTN_WAVE_LDS;
  const __bf16* qp  = Q  + ((size_t)b * T_DIM + t0) * C_DIM + h * D_DIM;
  const __bf16* kp  = Kb + (size_t)b * I_DIM * C_DIM + h * D_DIM;
  const __bf16* vtp = VT + ((size_t)b * H_DIM + h) * D_DIM * I_DIM;

  int m = lane & 15, hi = lane >> 4;

  // Pass 1: scores (Q already carries the 1/sqrt(D) scale)
  for (int nt = 0; nt < I_DIM / 16; ++nt) {
    v8f acc = {};
#pragma unroll
    for (int k0 = 0; k0 < D_DIM; k0 += 32) {
      v16bf a = load_a_frag(qp, C_DIM, k0, lane);
      v16bf kf = load_b_frag(kp + (size_t)(nt * 16) * C_DIM, C_DIM, k0, lane);
      acc = wmma_bf16(a, kf, acc);
    }
#pragma unroll
    for (int r = 0; r < 8; ++r) {
      lfloat* row = (lfloat*)(wbase + (r + 8 * hi) * ATTN_ROW_BYTES);
      row[nt * 16 + m] = acc[r];
    }
  }
  __syncthreads();

  // Softmax: lane l owns row l
  if (lane < 16) {
    lfloat* row  = (lfloat*)(wbase + lane * ATTN_ROW_BYTES);
    lbf16*  prow = (lbf16*)row;
    float mx = -3.0e38f;
    for (int i = 0; i < I_DIM; ++i) mx = fmaxf(mx, row[i]);
    float sum = 0.0f;
    for (int i = 0; i < I_DIM; ++i) { float e = __expf(row[i] - mx); row[i] = e; sum += e; }
    float rinv = 1.0f / sum;
    // bf16 write at byte 2i never clobbers an unread f32 at byte 4i (i asc.)
    for (int i = 0; i < I_DIM; ++i) prow[i] = (__bf16)(row[i] * rinv);
  }
  __syncthreads();

  // Pass 2: Y = P(16x576) * V(576x64)
  v8f yacc[4] = {};
  const lbf16* pr = (const lbf16*)(wbase + m * ATTN_ROW_BYTES);
  for (int kk = 0; kk < I_DIM / 32; ++kk) {
    int i0 = kk * 32;
    union { v16bf v; v8bf hh[2]; } ua;
    ua.hh[0] = *(const AS3 v8bf*)(pr + i0 + hi * 16);
    ua.hh[1] = *(const AS3 v8bf*)(pr + i0 + 8 + hi * 16);
#pragma unroll
    for (int j = 0; j < 4; ++j) {
      v16bf vb = *(const v16bf*)(vtp + (size_t)(j * 16 + m) * I_DIM + i0 + hi * 16);
      yacc[j] = wmma_bf16(ua.v, vb, yacc[j]);
    }
  }
#pragma unroll
  for (int j = 0; j < 4; ++j)
#pragma unroll
    for (int r = 0; r < 8; ++r) {
      int t = t0 + r + 8 * hi, d = j * 16 + m;
      Y[((size_t)b * T_DIM + t) * C_DIM + h * D_DIM + d] = (__bf16)yacc[j][r];
    }
}

// ---- GEMM 3: out = Y * Wo^T + bo (f32 output) -----------------------------
__global__ void __launch_bounds__(256) gemm_o(const __bf16* __restrict__ Yb,
                                              const __bf16* __restrict__ Wo,
                                              const float* __restrict__ bo,
                                              float* __restrict__ Out) {
  extern __shared__ __align__(16) char smem[];
  lchar* ls = (lchar*)smem;
  int lane = threadIdx.x & 31, wave = threadIdx.x >> 5;
  int wm = wave & 1, wn = wave >> 1;
  int row0 = blockIdx.x * 64, n0 = blockIdx.y * 256;
  v8f acc[2][4] = {};
  gemm_block<C_DIM>(Yb + (size_t)row0 * C_DIM, Wo + (size_t)n0 * C_DIM, ls, acc);
  int m = lane & 15, hi = lane >> 4;
#pragma unroll
  for (int mi = 0; mi < 2; ++mi)
#pragma unroll
    for (int j = 0; j < 4; ++j) {
      int col = n0 + wn * 64 + j * 16 + m;
      int rowb = row0 + wm * 32 + mi * 16;
      float bv = bo[col];
#pragma unroll
      for (int r = 0; r < 8; ++r)
        Out[(size_t)(rowb + r + 8 * hi) * C_DIM + col] = acc[mi][j][r] + bv;
    }
}

// ---------------------------------------------------------------------------
extern "C" void kernel_launch(void* const* d_in, const int* in_sizes, int n_in,
                              void* d_out, int out_size, void* d_ws, size_t ws_size,
                              hipStream_t stream) {
  const float* x   = (const float*)d_in[0];
  const float* enc = (const float*)d_in[1];
  const float* Wq  = (const float*)d_in[2];
  const float* bq  = (const float*)d_in[3];
  const float* Wkv = (const float*)d_in[4];
  const float* bkv = (const float*)d_in[5];
  const float* Wo  = (const float*)d_in[6];
  const float* bo  = (const float*)d_in[7];
  float* out = (float*)d_out;

  size_t off = 0;
  auto alloc = [&](size_t bytes) -> void* {
    void* p = (char*)d_ws + off;
    off += (bytes + 255) & ~(size_t)255;
    return p;
  };
  const size_t nX = (size_t)B_DIM * T_DIM * C_DIM;   // 8388608
  const size_t nE = (size_t)B_DIM * I_DIM * C_DIM;   // 4718592
  __bf16* xbf  = (__bf16*)alloc(nX * 2);
  __bf16* ebf  = (__bf16*)alloc(nE * 2);
  __bf16* wqb  = (__bf16*)alloc((size_t)C_DIM * C_DIM * 2);
  __bf16* wkvb = (__bf16*)alloc((size_t)2 * C_DIM * C_DIM * 2);
  __bf16* wob  = (__bf16*)alloc((size_t)C_DIM * C_DIM * 2);
  __bf16* qb   = (__bf16*)alloc(nX * 2);
  __bf16* kb   = (__bf16*)alloc(nE * 2);
  __bf16* vtb  = (__bf16*)alloc(nE * 2);
  __bf16* yb   = (__bf16*)alloc(nX * 2);

  // Stage 0: fp32 -> bf16
  cvt_bf16<<<(int)((nX + 255) / 256), 256, 0, stream>>>(x, xbf, (int)nX);
  cvt_bf16<<<(int)((nE + 255) / 256), 256, 0, stream>>>(enc, ebf, (int)nE);
  cvt_bf16<<<(C_DIM * C_DIM + 255) / 256, 256, 0, stream>>>(Wq, wqb, C_DIM * C_DIM);
  cvt_bf16<<<(2 * C_DIM * C_DIM + 255) / 256, 256, 0, stream>>>(Wkv, wkvb, 2 * C_DIM * C_DIM);
  cvt_bf16<<<(C_DIM * C_DIM + 255) / 256, 256, 0, stream>>>(Wo, wob, C_DIM * C_DIM);

  // Stage 1: Q projection (1/sqrt(D) folded in)
  gemm_q<<<dim3((B_DIM * T_DIM) / 64, C_DIM / 256), 256, GEMM_LDS_BYTES, stream>>>(
      xbf, wqb, bq, qb);
  // Stage 2: KV projection (V stored transposed per head)
  gemm_kv<<<dim3((B_DIM * I_DIM) / 64, (2 * C_DIM) / 256), 256, GEMM_LDS_BYTES, stream>>>(
      ebf, wkvb, bkv, kb, vtb);
  // Stage 3: attention (8192 wave tiles, 2 waves/block)
  attn<<<4096, 64, 2 * ATTN_WAVE_LDS, stream>>>(qb, kb, vtb, yb);
  // Stage 4: output projection -> f32
  gemm_o<<<dim3((B_DIM * T_DIM) / 64, C_DIM / 256), 256, GEMM_LDS_BYTES, stream>>>(
      yb, wob, bo, out);
}
